// MultiHeadAttention_4810363372374
// MI455X (gfx1250) — matrix operand order repaired and checked by reading, not compiled, not run on hardware
//
#include <hip/hip_runtime.h>
#include <hip/hip_bf16.h>

typedef __attribute__((ext_vector_type(16))) _Float16 v16h;
typedef __attribute__((ext_vector_type(8)))  float    v8f;

// Problem constants (fixed by the reference)
constexpr int Bb = 2, Ss = 2048, Dd = 1024, Hh = 16, HD = 64;
constexpr int Mrows = Bb * Ss;            // 4096
constexpr int NE_X  = Mrows * Dd;         // 4,194,304 elements
constexpr int NE_W  = Dd * Dd;            // 1,048,576 elements

// ---------------- WMMA helpers ----------------

static __device__ __forceinline__ v8f wmma16(v16h a, v16h b, v8f c) {
  // D = A(16x32 f16) * B(32x16 f16) + C(16x16 f32)
  return __builtin_amdgcn_wmma_f32_16x16x32_f16(
      /*neg_a=*/false, a, /*neg_b=*/false, b,
      /*c_mod=*/(short)0, c, /*reuse_a=*/false, /*reuse_b=*/false);
}

// A-operand loader: 16x32 f16 tile, row-major source with leading dim `ld`.
// Contiguous-k layout used here (not AMD's table order): lane row = lane&15; VGPR v holds K pair
//   kb = (lane>=16?16:0) + v*2
static __device__ __forceinline__ v16h ld_a(const _Float16* base, int ld) {
  int l = threadIdx.x & 31, r = l & 15, hi = l >> 4;
  union { v16h v; _Float16 h[16]; } u;
  const _Float16* rp = base + r * ld;
#pragma unroll
  for (int i = 0; i < 8; ++i) {
    int kb = (hi << 4) + (i << 1);
    u.h[2 * i]     = rp[kb];
    u.h[2 * i + 1] = rp[kb + 1];
  }
  return u.v;
}

// B-operand loader for B = R^T, where R is row-major [16 x 32] with leading
// dim `ld` (so B(k,n) = R[n][k]).  Contiguous-k layout: lane column n = lane&15;
// VGPR v holds K pair k = (lane>=16?16:0) + 2v,{+1}  -> 32B contiguous/lane.
static __device__ __forceinline__ v16h ld_bT(const _Float16* base, int ld) {
  int l = threadIdx.x & 31, n = l & 15, hi = l >> 4;
  union { v16h v; _Float16 h[16]; } u;
  const _Float16* rp = base + n * ld + (hi << 4);
#pragma unroll
  for (int i = 0; i < 8; ++i) {
    u.h[2 * i]     = rp[2 * i];
    u.h[2 * i + 1] = rp[2 * i + 1];
  }
  return u.v;
}

// ---------------- CDNA5 async global->LDS (ASYNCcnt) ----------------

static __device__ __forceinline__ void async_ld_b128(unsigned lds_off, const void* gptr) {
  // VGLOBAL async: VDST = LDS byte address, VADDR = 64-bit global address (GV mode)
  asm volatile("global_load_async_to_lds_b128 %0, %1, off"
               :: "v"(lds_off), "v"((unsigned long long)(uintptr_t)gptr)
               : "memory");
}
static __device__ __forceinline__ void wait_async0() {
  asm volatile("s_wait_asynccnt 0x0" ::: "memory");
}
static __device__ __forceinline__ void wait_async1() {
  asm volatile("s_wait_asynccnt 0x1" ::: "memory");
}

// ---------------- conversion kernels ----------------

__global__ __launch_bounds__(256) void cvt_f32_to_f16(const float* __restrict__ in,
                                                      _Float16* __restrict__ out, int n) {
  int i = blockIdx.x * 256 + threadIdx.x;
  if (i < n) out[i] = (_Float16)in[i];
}

// Weights W[k][n] (row-major DxD) -> WT[n][k] f16 (row-major), so the WMMA
// B-operand K-pairs are contiguous loads.
__global__ __launch_bounds__(256) void cvtT_f32_to_f16(const float* __restrict__ in,
                                                       _Float16* __restrict__ out) {
  int i = blockIdx.x * 256 + threadIdx.x;   // 0 .. D*D-1
  int n = i >> 10, k = i & 1023;            // D == 1024
  out[i] = (_Float16)in[k * Dd + n];
}

// ---------------- WMMA GEMM:  C = A[MxK] * BT^T[KxN] ----------------
// A f16 row-major (streamed from global, next-step fragment prefetched),
// BT f16 row-major [N x K], staged per-block into LDS with double-buffered
// ASYNC global->LDS copies (one b128 per thread per panel).
// MODE 0: f16 row-major out.  MODE 1: f16 out scattered as V^T[b,h,hd,s].
// MODE 2: f32 row-major out.
// Block = 256 threads = 8 waves; block tile 128(M) x 64(N); wave tile 16 x 64.

template <int MODE>
__global__ __launch_bounds__(256, 2)
void gemm_wmma(const _Float16* __restrict__ A, const _Float16* __restrict__ BT,
               void* __restrict__ Cout, int M, int N, int K) {
  __shared__ _Float16 bpanel[2][64 * 32];   // double-buffered 4KB B panels
  int tid  = threadIdx.x;
  int lane = tid & 31, wave = tid >> 5;
  int bm = blockIdx.y * 128 + wave * 16;
  int bn = blockIdx.x * 64;

  // stage BT rows [bn..bn+63] x k[k0..k0+31] -> bpanel[buf] (row-major, ld=32)
  auto stage = [&](int k0, int buf) {
    int r = tid >> 2, koff = (tid & 3) * 8;            // 16B per thread
    const _Float16* g = BT + (size_t)(bn + r) * K + k0 + koff;
    unsigned l = (unsigned)(uintptr_t)(&bpanel[buf][0]) + (unsigned)tid * 16u;
    async_ld_b128(l, g);
  };

  stage(0, 0);
  v16h a = ld_a(A + (size_t)bm * K, K);
  v8f acc[4] = {};
  int steps = K / 32;

  for (int t = 0; t < steps; ++t) {
    if (t + 1 < steps) { stage((t + 1) * 32, (t + 1) & 1); wait_async1(); }
    else               { wait_async0(); }
    __syncthreads();                      // panel t visible to all waves

    v16h a_next = a;
    if (t + 1 < steps)                    // prefetch next A fragment early
      a_next = ld_a(A + (size_t)bm * K + (t + 1) * 32, K);

    const _Float16* bp = &bpanel[t & 1][0];
#pragma unroll
    for (int c = 0; c < 4; ++c) {
      v16h b = ld_bT(bp + (c * 16) * 32, 32);   // ds_load, 32B/lane contiguous
      acc[c] = wmma16(a, b, acc[c]);
    }
    a = a_next;
    __syncthreads();                      // all reads of panel t done
  }

  int hi = lane >> 4, nl = lane & 15;
#pragma unroll
  for (int c = 0; c < 4; ++c) {
#pragma unroll
    for (int v = 0; v < 8; ++v) {
      int row = bm + v + 8 * hi;           // M index (lane half selects +8)
      int col = bn + c * 16 + nl;          // N index
      float val = acc[c][v];
      if (MODE == 0) {
        ((_Float16*)Cout)[(size_t)row * N + col] = (_Float16)val;
      } else if (MODE == 1) {
        // scatter as V^T[b][h][hd][s]
        int b_ = row >> 11, s_ = row & 2047;   // S == 2048
        int h_ = col >> 6,  d_ = col & 63;     // HD == 64
        ((_Float16*)Cout)[((size_t)((b_ * Hh + h_) * HD + d_)) * Ss + s_] = (_Float16)val;
      } else {
        ((float*)Cout)[(size_t)row * N + col] = val;
      }
    }
  }
}

// ---------------- fused causal flash attention ----------------
// One wave per (b, h, 16-query tile).  Computes S^T = K * Q^T so softmax
// reductions are per-lane (over VGPRs) + one half-wave shuffle; probabilities
// staged via LDS to re-layout into the B operand of O^T = V^T * P^T.

__global__ __launch_bounds__(128, 2)
void flash_fwd(const _Float16* __restrict__ Q, const _Float16* __restrict__ Kk,
               const _Float16* __restrict__ VT, _Float16* __restrict__ O) {
  __shared__ _Float16 lds[4 * 16 * 32];     // 4 waves x (16 q x 32 keys), 4 KB
  int lane = threadIdx.x & 31, wid = threadIdx.x >> 5;
  int w  = blockIdx.x * 4 + wid;            // global wave id: 4096 total
  int qt = w & 127;                         // S/16 = 128 query tiles
  int h  = (w >> 7) & 15;
  int b  = w >> 11;
  int nl = lane & 15, hi = lane >> 4;
  int qg = qt * 16 + nl;                    // this lane's query row

  // Q^T B-operand fragments (resident for the whole key loop), kdim 0..31 / 32..63
  const _Float16* qbase = Q + ((size_t)(b * Ss + qt * 16)) * Dd + h * HD;
  v16h bq0 = ld_bT(qbase, Dd);
  v16h bq1 = ld_bT(qbase + 32, Dd);

  float mi = -INFINITY, si = 0.f;
  v8f o[4] = {};
  _Float16* myl = lds + wid * 512;

  int ntiles = (qt * 16 + 15) / 32 + 1;     // causal: only up to diagonal
  for (int t = 0; t < ntiles; ++t) {
    int kb = t * 32;
    float sarr[16];
    float mloc = -INFINITY;
#pragma unroll
    for (int sub = 0; sub < 2; ++sub) {
      // S^T(key, q) for 16 keys: A = K rows, two WMMAs over hd
      const _Float16* kbase = Kk + ((size_t)(b * Ss + kb + sub * 16)) * Dd + h * HD;
      v16h a0 = ld_a(kbase, Dd);
      v16h a1 = ld_a(kbase + 32, Dd);
      v8f s4 = {};
      s4 = wmma16(a0, bq0, s4);
      s4 = wmma16(a1, bq1, s4);
#pragma unroll
      for (int v = 0; v < 8; ++v) {
        int key = kb + sub * 16 + v + 8 * hi;
        float sv = s4[v] * 0.125f;               // 1/sqrt(64)
        sv = (key > qg) ? -1.0e30f : sv;         // causal mask
        sarr[sub * 8 + v] = sv;
        mloc = fmaxf(mloc, sv);
      }
    }
    // per-query stats: combine the two lane halves holding the same query
    mloc = fmaxf(mloc, __shfl_xor(mloc, 16, 32));
    float mnew  = fmaxf(mi, mloc);
    float alpha = __expf(mi - mnew);
    float rs = 0.f;
#pragma unroll
    for (int i2 = 0; i2 < 16; ++i2) {
      float p = __expf(sarr[i2] - mnew);
      sarr[i2] = p;
      rs += p;
    }
    rs += __shfl_xor(rs, 16, 32);
    si = si * alpha + rs;
    mi = mnew;
#pragma unroll
    for (int c = 0; c < 4; ++c) o[c] = o[c] * alpha;

    // stage P as LDS[q][key] (f16) to re-layout D-regs -> B-operand
#pragma unroll
    for (int sub = 0; sub < 2; ++sub) {
#pragma unroll
      for (int v = 0; v < 8; ++v) {
        int keyl = sub * 16 + v + 8 * hi;
        myl[nl * 32 + keyl] = (_Float16)sarr[sub * 8 + v];
      }
    }
    __asm__ volatile("s_wait_dscnt 0" ::: "memory");
    v16h bp = ld_bT(myl, 32);                // B = P^T (32 keys x 16 q)

    // O^T += V^T * P^T : 4 hd-chunks of 16 rows
#pragma unroll
    for (int c = 0; c < 4; ++c) {
      v16h av = ld_a(VT + ((size_t)((b * Hh + h) * HD + c * 16)) * Ss + kb, Ss);
      o[c] = wmma16(av, bp, o[c]);
    }
  }

  // normalize and store attention output (f16, row-major [B*S, D])
  float inv = 1.0f / si;
#pragma unroll
  for (int c = 0; c < 4; ++c) {
#pragma unroll
    for (int v = 0; v < 8; ++v) {
      int hd = c * 16 + v + 8 * hi;
      O[((size_t)(b * Ss + qg)) * Dd + h * HD + hd] = (_Float16)(o[c][v] * inv);
    }
  }
}

// ---------------- host-side launch ----------------

extern "C" void kernel_launch(void* const* d_in, const int* in_sizes, int n_in,
                              void* d_out, int out_size, void* d_ws, size_t ws_size,
                              hipStream_t stream) {
  (void)in_sizes; (void)n_in; (void)out_size; (void)ws_size;
  const float* x  = (const float*)d_in[0];
  const float* WQ = (const float*)d_in[1];
  const float* WK = (const float*)d_in[2];
  const float* WV = (const float*)d_in[3];
  const float* WO = (const float*)d_in[4];

  // f16 workspace layout (elements)
  _Float16* ws  = (_Float16*)d_ws;
  _Float16* xh  = ws;                       // [4096 x 1024]
  _Float16* qh  = ws + (size_t)NE_X;        // [4096 x 1024]
  _Float16* kh  = ws + (size_t)NE_X * 2;    // [4096 x 1024]
  _Float16* vth = ws + (size_t)NE_X * 3;    // V^T [2,16,64,2048]
  _Float16* ah  = ws + (size_t)NE_X * 4;    // attn out [4096 x 1024]
  _Float16* wqT = ws + (size_t)NE_X * 5;
  _Float16* wkT = wqT + NE_W;
  _Float16* wvT = wkT + NE_W;
  _Float16* woT = wvT + NE_W;

  // 1) convert activations + transposed weights to f16
  cvt_f32_to_f16<<<NE_X / 256, 256, 0, stream>>>(x, xh, NE_X);
  cvtT_f32_to_f16<<<NE_W / 256, 256, 0, stream>>>(WQ, wqT);
  cvtT_f32_to_f16<<<NE_W / 256, 256, 0, stream>>>(WK, wkT);
  cvtT_f32_to_f16<<<NE_W / 256, 256, 0, stream>>>(WV, wvT);
  cvtT_f32_to_f16<<<NE_W / 256, 256, 0, stream>>>(WO, woT);

  // 2) QKV projections (WMMA + async-LDS B panels), V scattered transposed
  dim3 g(Dd / 64, Mrows / 128);             // (16, 32)
  gemm_wmma<0><<<g, 256, 0, stream>>>(xh, wqT, qh,  Mrows, Dd, Dd);
  gemm_wmma<0><<<g, 256, 0, stream>>>(xh, wkT, kh,  Mrows, Dd, Dd);
  gemm_wmma<1><<<g, 256, 0, stream>>>(xh, wvT, vth, Mrows, Dd, Dd);

  // 3) fused causal flash attention: 4096 waves, 4 waves/block
  flash_fwd<<<1024, 128, 0, stream>>>(qh, kh, vth, ah);

  // 4) output projection, fp32 result into d_out
  gemm_wmma<2><<<g, 256, 0, stream>>>(ah, woT, d_out, Mrows, Dd, Dd);
}
